// BiRNN_CRF_4097398800495
// MI455X (gfx1250) — matrix operand order repaired and checked by reading, not compiled, not run on hardware
//
#include <hip/hip_runtime.h>
#include <hip/hip_bf16.h>
#include <math.h>

// ---------------------------------------------------------------------------
// BiLSTM-CRF forward (NLL) for MI455X / gfx1250.
//  - bf16 WMMA (v_wmma_f32_16x16x32_bf16) for all GEMMs, fp32 accumulate.
//  - Input-side LSTM GEMMs batched over all S*B rows.
//  - Recurrent part: one persistent workgroup (32 waves) per direction,
//    h double-buffered in LDS, c in VGPRs, Whh streamed bf16 from L2.
//  - CRF numerator + forward algorithm in a single small block (fp32).
// ---------------------------------------------------------------------------

typedef __attribute__((ext_vector_type(16))) __bf16 v16bf;
typedef __attribute__((ext_vector_type(8)))  float  v8f;

union Frag32B { v16bf v; uint4 q[2]; };

#define WMMA_BF16(A_, B_, C_) \
  __builtin_amdgcn_wmma_f32_16x16x32_bf16(false, (A_), false, (B_), (short)0, (C_), false, false)

// A fragment: 16x32 (MxK) bf16, source row-major with leading dim lda.
// Contiguous K order used here (same as B): lane holds row M=lane&15; halves e=0..15 -> K = k0 + e
// with k0 = kk + (lane>=16 ? 16 : 0).  => two contiguous 16B loads.
__device__ __forceinline__ v16bf load_frag_a(const __bf16* A, int lda, int rowBase,
                                             int kk, int lane) {
  const int row = rowBase + (lane & 15);
  const int k0  = kk + ((lane >> 4) << 4);
  const __bf16* p = A + (size_t)row * lda + k0;
  Frag32B f;
  f.q[0] = *reinterpret_cast<const uint4*>(p);
  f.q[1] = *reinterpret_cast<const uint4*>(p + 8);
  return f.v;
}

// B fragment: 32x16 (KxN) bf16, weight stored as [N][K] row-major (PyTorch W).
// Contiguous K order used here (same as A): lane holds column N=lane&15; halves e=0..15 -> K = kk + (lane>=16?16:0) + e.
// => 16 consecutive K per lane = two contiguous 16B loads.
__device__ __forceinline__ v16bf load_frag_b(const __bf16* W, int ldw, int colBase,
                                             int kk, int lane) {
  const int n  = colBase + (lane & 15);
  const int k0 = kk + ((lane >> 4) << 4);
  const __bf16* p = W + (size_t)n * ldw + k0;
  Frag32B f;
  f.q[0] = *reinterpret_cast<const uint4*>(p);
  f.q[1] = *reinterpret_cast<const uint4*>(p + 8);
  return f.v;
}

__device__ __forceinline__ float sigm(float x) { return 1.f / (1.f + expf(-x)); }

// ---------------------------------------------------------------------------
// Elementwise prep kernels
// ---------------------------------------------------------------------------

// fp32 [srows,scols] -> bf16 [drows,dcols], zero-padded.
__global__ void convert_pad_kernel(const float* __restrict__ src, __bf16* __restrict__ dst,
                                   int srows, int scols, int drows, int dcols) {
  size_t n = (size_t)drows * dcols;
  for (size_t i = (size_t)blockIdx.x * blockDim.x + threadIdx.x; i < n;
       i += (size_t)gridDim.x * blockDim.x) {
    int r = (int)(i / dcols), c = (int)(i % dcols);
    float v = (r < srows && c < scols) ? src[(size_t)r * scols + c] : 0.f;
    dst[i] = (__bf16)v;
  }
}

__global__ void bias_comb_kernel(const float* __restrict__ a, const float* __restrict__ b,
                                 float* __restrict__ o, int n) {
  for (int i = blockIdx.x * blockDim.x + threadIdx.x; i < n; i += gridDim.x * blockDim.x)
    o[i] = a[i] + b[i];
}

__global__ void bias_pad_kernel(const float* __restrict__ a, float* __restrict__ o,
                                int nsrc, int ndst) {
  for (int i = blockIdx.x * blockDim.x + threadIdx.x; i < ndst; i += gridDim.x * blockDim.x)
    o[i] = (i < nsrc) ? a[i] : 0.f;
}

// Embedding gather -> time-major bf16 X0 [S][B][ld], cols >= E zero.
__global__ void embed_kernel(const int* __restrict__ ids, const float* __restrict__ table,
                             __bf16* __restrict__ X, int S, int B, int E, int ld) {
  size_t n = (size_t)S * B * ld;
  for (size_t i = (size_t)blockIdx.x * blockDim.x + threadIdx.x; i < n;
       i += (size_t)gridDim.x * blockDim.x) {
    int c = (int)(i % ld);
    size_t sb = i / ld;
    int b = (int)(sb % B), s = (int)(sb / B);
    float v = 0.f;
    if (c < E) v = table[(size_t)ids[(size_t)b * S + s] * E + c];
    X[i] = (__bf16)v;
  }
}

// ---------------------------------------------------------------------------
// Generic bf16 GEMM: C[M][N] = A[M][K](bf16,row-major) * W[N][K]^T + bias[N]
// Block = 256 threads = 8 waves; WG tile 128x64, wave tile 32x32 (2x2 WMMA).
// Grid: (N/64, M/128).  M%128==0, N%64==0, K%32==0 guaranteed by caller.
// ---------------------------------------------------------------------------
__global__ __launch_bounds__(256) void gemm_bf16_kernel(
    const __bf16* __restrict__ A, const __bf16* __restrict__ W,
    const float* __restrict__ bias, float* __restrict__ C,
    int K, int lda, int ldw, int ldc) {
  const int lane = threadIdx.x & 31;
  const int ww   = threadIdx.x >> 5;
  const int Mw = blockIdx.y * 128 + (ww >> 1) * 32;
  const int Nw = blockIdx.x * 64 + (ww & 1) * 32;

  const v8f z = {0.f, 0.f, 0.f, 0.f, 0.f, 0.f, 0.f, 0.f};
  v8f acc00 = z, acc01 = z, acc10 = z, acc11 = z;

  for (int kk = 0; kk < K; kk += 32) {
    if (kk + 64 < K) {  // uniform; speculative prefetch of next-next A slab
      __builtin_prefetch(A + (size_t)(Mw + (lane & 15)) * lda + kk + 64, 0, 0);
    }
    v16bf a0 = load_frag_a(A, lda, Mw, kk, lane);
    v16bf a1 = load_frag_a(A, lda, Mw + 16, kk, lane);
    v16bf b0 = load_frag_b(W, ldw, Nw, kk, lane);
    v16bf b1 = load_frag_b(W, ldw, Nw + 16, kk, lane);
    acc00 = WMMA_BF16(a0, b0, acc00);
    acc01 = WMMA_BF16(a0, b1, acc01);
    acc10 = WMMA_BF16(a1, b0, acc10);
    acc11 = WMMA_BF16(a1, b1, acc11);
  }

  const int hi8 = (lane >> 4) << 3;
  const int nlo = lane & 15;
#pragma unroll
  for (int r = 0; r < 8; ++r) {
    int row0 = Mw + hi8 + r, row1 = Mw + 16 + hi8 + r;
    int col0 = Nw + nlo, col1 = Nw + 16 + nlo;
    C[(size_t)row0 * ldc + col0] = acc00[r] + bias[col0];
    C[(size_t)row0 * ldc + col1] = acc01[r] + bias[col1];
    C[(size_t)row1 * ldc + col0] = acc10[r] + bias[col0];
    C[(size_t)row1 * ldc + col1] = acc11[r] + bias[col1];
  }
}

// ---------------------------------------------------------------------------
// Persistent LSTM recurrence. grid=2 (blockIdx.x = direction), block=1024
// (32 waves).  Wave w owns h columns [16w,16w+16): 8 WMMA tiles/step
// (2 M-halves x 4 gates), K=512 -> 128 v_wmma per wave per step.
// gates_* : [S][B][2048] fp32, already = x@Wih^T + bih + bhh.
// whh_*   : [2048][512] bf16 (original [4H][H] row-major).
// hout    : [S][B][1024] bf16, fwd in cols 0..511, bwd in 512..1023.
// ---------------------------------------------------------------------------
__global__ __launch_bounds__(1024) void lstm_scan_kernel(
    const float* __restrict__ gates_f, const float* __restrict__ gates_b,
    const __bf16* __restrict__ whh_f, const __bf16* __restrict__ whh_b,
    __bf16* __restrict__ hout) {
  const int dir = blockIdx.x;
  const float* gates = dir ? gates_b : gates_f;
  const __bf16* whh  = dir ? whh_b : whh_f;
  const int dirOff   = dir ? 512 : 0;

  __shared__ __attribute__((aligned(16))) __bf16 hbuf[2][32][512];  // 64 KB, double-buffered h

  const int lane = threadIdx.x & 31;
  const int w    = threadIdx.x >> 5;  // 0..31: h-column block
  const int hi8  = (lane >> 4) << 3;
  const int nlo  = lane & 15;

  // zero h0
  {
    __bf16* hz = &hbuf[0][0][0];
    for (int i = threadIdx.x; i < 32 * 512; i += 1024) hz[i] = (__bf16)0.f;
  }
  float creg[2][8];
#pragma unroll
  for (int m = 0; m < 2; ++m)
#pragma unroll
    for (int r = 0; r < 8; ++r) creg[m][r] = 0.f;
  __syncthreads();

  for (int t = 0; t < 512; ++t) {
    const int cur = t & 1, nxt = cur ^ 1;
    const int s = dir ? (511 - t) : t;
    const float* gs = gates + (size_t)s * (32 * 2048);

    // init accumulators with input-side gate pre-activations (C-fragment layout)
    v8f acc[2][4];
#pragma unroll
    for (int m = 0; m < 2; ++m)
#pragma unroll
      for (int g = 0; g < 4; ++g)
#pragma unroll
        for (int r = 0; r < 8; ++r)
          acc[m][g][r] = gs[(size_t)(16 * m + hi8 + r) * 2048 + (512 * g + 16 * w + nlo)];

    // recurrent GEMM: h_prev (LDS) x Whh^T
    const __bf16* hp = &hbuf[cur][0][0];
    for (int kk = 0; kk < 512; kk += 32) {
      v16bf a0 = load_frag_a(hp, 512, 0, kk, lane);
      v16bf a1 = load_frag_a(hp, 512, 16, kk, lane);
#pragma unroll
      for (int g = 0; g < 4; ++g) {
        v16bf bg = load_frag_b(whh, 512, 512 * g + 16 * w, kk, lane);
        acc[0][g] = WMMA_BF16(a0, bg, acc[0][g]);
        acc[1][g] = WMMA_BF16(a1, bg, acc[1][g]);
      }
    }

    // gate nonlinearities, state update, write h (LDS + global)
#pragma unroll
    for (int m = 0; m < 2; ++m)
#pragma unroll
      for (int r = 0; r < 8; ++r) {
        float gi = acc[m][0][r], gf = acc[m][1][r], gg = acc[m][2][r], go = acc[m][3][r];
        float c = sigm(gf) * creg[m][r] + sigm(gi) * tanhf(gg);
        creg[m][r] = c;
        float h = sigm(go) * tanhf(c);
        int b = 16 * m + hi8 + r;
        int col = 16 * w + nlo;
        __bf16 hb = (__bf16)h;
        hbuf[nxt][b][col] = hb;
        hout[((size_t)s * 32 + b) * 1024 + dirOff + col] = hb;
      }
    __syncthreads();
  }
}

// ---------------------------------------------------------------------------
// CRF token-mean NLL.  em: [S][B][64] fp32 (tags 0..8 valid). One block, 512 thr.
// ---------------------------------------------------------------------------
__global__ __launch_bounds__(512) void crf_kernel(
    const float* __restrict__ em, const int* __restrict__ tags,
    const int* __restrict__ mask, const float* __restrict__ start,
    const float* __restrict__ endt, const float* __restrict__ trans,
    float* __restrict__ out) {
  const int S = 512, T = 9;
  __shared__ float str[9], etr[9], tr[9][9];
  __shared__ float score[32][16];
  __shared__ float numb[32], denomb[32];
  __shared__ int cnt[32];

  int tid = threadIdx.x;
  if (tid < T) { str[tid] = start[tid]; etr[tid] = endt[tid]; }
  if (tid < T * T) tr[tid / T][tid % T] = trans[tid];
  __syncthreads();

#define EMV(s_, b_, t_) em[(((size_t)(s_)) * 32 + (b_)) * 64 + (t_)]

  // numerator (path score) per batch, serial over S
  if (tid < 32) {
    int b = tid;
    int t0 = tags[(size_t)b * S];
    float num = str[t0] + EMV(0, b, t0);
    int count = (mask[(size_t)b * S] != 0) ? 1 : 0;
    for (int s = 1; s < S; ++s) {
      int m = mask[(size_t)b * S + s];
      int tp = tags[(size_t)b * S + s - 1];
      int tc = tags[(size_t)b * S + s];
      if (m) { num += tr[tp][tc] + EMV(s, b, tc); count++; }
    }
    int last = tags[(size_t)b * S + (count - 1)];
    num += etr[last];
    numb[b] = num;
    cnt[b] = count;
  }
  __syncthreads();

  // forward algorithm (log partition)
  int b = tid >> 4, j = tid & 15;
  if (j < T) score[b][j] = str[j] + EMV(0, b, j);
  __syncthreads();
  for (int s = 1; s < S; ++s) {
    float nv = 0.f;
    if (j < T) {
      float mx = -INFINITY;
      for (int i = 0; i < T; ++i) mx = fmaxf(mx, score[b][i] + tr[i][j]);
      float sum = 0.f;
      for (int i = 0; i < T; ++i) sum += expf(score[b][i] + tr[i][j] - mx);
      nv = mx + logf(sum) + EMV(s, b, j);
      if (!mask[(size_t)b * S + s]) nv = score[b][j];
    }
    __syncthreads();
    if (j < T) score[b][j] = nv;
    __syncthreads();
  }
  if (j == 0) {
    float mx = -INFINITY;
    for (int i = 0; i < T; ++i) mx = fmaxf(mx, score[b][i] + etr[i]);
    float sum = 0.f;
    for (int i = 0; i < T; ++i) sum += expf(score[b][i] + etr[i] - mx);
    denomb[b] = mx + logf(sum);
  }
  __syncthreads();
  if (tid == 0) {
    float ll = 0.f, mtot = 0.f;
    for (int bb = 0; bb < 32; ++bb) { ll += numb[bb] - denomb[bb]; mtot += (float)cnt[bb]; }
    out[0] = -(ll / mtot);
  }
#undef EMV
}

// ---------------------------------------------------------------------------
// Host-side orchestration
// ---------------------------------------------------------------------------
extern "C" void kernel_launch(void* const* d_in, const int* in_sizes, int n_in,
                              void* d_out, int out_size, void* d_ws, size_t ws_size,
                              hipStream_t stream) {
  (void)in_sizes; (void)n_in; (void)out_size; (void)ws_size;
  const int B = 32, S = 512, E = 300, H = 512, T = 9;
  const int M = S * B;       // 16384
  const int G = 4 * H;       // 2048

  const int*   ids   = (const int*)d_in[0];
  const int*   mask  = (const int*)d_in[1];
  const int*   tags  = (const int*)d_in[3];
  const float* table = (const float*)d_in[4];
  const float *Wih[2][2], *Whh[2][2], *bih[2][2], *bhh[2][2];
  int p = 5;
  for (int layer = 0; layer < 2; ++layer)
    for (int dir = 0; dir < 2; ++dir) {
      Wih[layer][dir] = (const float*)d_in[p++];
      Whh[layer][dir] = (const float*)d_in[p++];
      bih[layer][dir] = (const float*)d_in[p++];
      bhh[layer][dir] = (const float*)d_in[p++];
    }
  const float* W_tag  = (const float*)d_in[21];
  const float* b_tag  = (const float*)d_in[22];
  const float* startp = (const float*)d_in[23];
  const float* endp   = (const float*)d_in[24];
  const float* transp = (const float*)d_in[25];

  // workspace carve-up (256B aligned)
  char* wsb = (char*)d_ws;
  size_t off = 0;
  auto alloc = [&](size_t bytes) -> void* {
    off = (off + 255) & ~(size_t)255;
    void* q = wsb + off;
    off += bytes;
    return q;
  };

  __bf16* WihP[2][2];
  __bf16* WhhB[2][2];
  float*  biasC[2][2];
  for (int layer = 0; layer < 2; ++layer)
    for (int dir = 0; dir < 2; ++dir) {
      int kin = (layer == 0) ? 320 : 1024;            // padded K
      WihP[layer][dir]  = (__bf16*)alloc((size_t)G * kin * 2);
      WhhB[layer][dir]  = (__bf16*)alloc((size_t)G * H * 2);
      biasC[layer][dir] = (float*)alloc((size_t)G * 4);
    }
  __bf16* WtagP = (__bf16*)alloc((size_t)64 * 1024 * 2);
  float*  btagP = (float*)alloc(64 * 4);
  __bf16* X0 = (__bf16*)alloc((size_t)M * 320 * 2);
  __bf16* X1 = (__bf16*)alloc((size_t)M * 1024 * 2);
  __bf16* H2 = (__bf16*)alloc((size_t)M * 1024 * 2);
  float* gatesF = (float*)alloc((size_t)M * G * 4);
  float* gatesB = (float*)alloc((size_t)M * G * 4);
  float* emis   = (float*)alloc((size_t)M * 64 * 4);

  // --- weight conversion (fp32 -> bf16, pad) + bias combine ---
  for (int layer = 0; layer < 2; ++layer)
    for (int dir = 0; dir < 2; ++dir) {
      int kin_s = (layer == 0) ? E : 1024;
      int kin_d = (layer == 0) ? 320 : 1024;
      convert_pad_kernel<<<1024, 256, 0, stream>>>(Wih[layer][dir], WihP[layer][dir],
                                                   G, kin_s, G, kin_d);
      convert_pad_kernel<<<512, 256, 0, stream>>>(Whh[layer][dir], WhhB[layer][dir],
                                                  G, H, G, H);
      bias_comb_kernel<<<8, 256, 0, stream>>>(bih[layer][dir], bhh[layer][dir],
                                              biasC[layer][dir], G);
    }
  convert_pad_kernel<<<256, 256, 0, stream>>>(W_tag, WtagP, T, 2 * H, 64, 1024);
  bias_pad_kernel<<<1, 64, 0, stream>>>(b_tag, btagP, T, 64);

  // --- embedding gather (time-major, padded to K=320) ---
  embed_kernel<<<2048, 256, 0, stream>>>(ids, table, X0, S, B, E, 320);

  // --- layer 0: input GEMMs, then recurrent scan (fwd+bwd concurrently) ---
  dim3 blk(256);
  dim3 gG(G / 64, M / 128);
  gemm_bf16_kernel<<<gG, blk, 0, stream>>>(X0, WihP[0][0], biasC[0][0], gatesF,
                                           320, 320, 320, G);
  gemm_bf16_kernel<<<gG, blk, 0, stream>>>(X0, WihP[0][1], biasC[0][1], gatesB,
                                           320, 320, 320, G);
  lstm_scan_kernel<<<2, 1024, 0, stream>>>(gatesF, gatesB, WhhB[0][0], WhhB[0][1], X1);

  // --- layer 1 ---
  gemm_bf16_kernel<<<gG, blk, 0, stream>>>(X1, WihP[1][0], biasC[1][0], gatesF,
                                           1024, 1024, 1024, G);
  gemm_bf16_kernel<<<gG, blk, 0, stream>>>(X1, WihP[1][1], biasC[1][1], gatesB,
                                           1024, 1024, 1024, G);
  lstm_scan_kernel<<<2, 1024, 0, stream>>>(gatesF, gatesB, WhhB[1][0], WhhB[1][1], H2);

  // --- tag projection (N padded 9 -> 64) ---
  dim3 gT(1, M / 128);
  gemm_bf16_kernel<<<gT, blk, 0, stream>>>(H2, WtagP, btagP, emis, 1024, 1024, 1024, 64);

  // --- CRF NLL ---
  crf_kernel<<<1, 512, 0, stream>>>(emis, tags, mask, startp, endp, transp, (float*)d_out);
}
